// Decoder_50775103373422
// MI455X (gfx1250) — matrix operand order repaired and checked by reading, not compiled, not run on hardware
//
#include <hip/hip_runtime.h>
#include <stdint.h>

// ---------------------------------------------------------------------------
// Problem constants (from reference)
// ---------------------------------------------------------------------------
#define BB   64      // batch
#define TT   25      // timesteps
#define HH   2048    // hidden
#define G4H  8192    // 4*H (gates)
#define NIN  66
#define NOUT 66
#define KPAD 96      // IN padded to 3 k-tiles of 32

typedef __bf16 bf16_t;
typedef __attribute__((ext_vector_type(16))) __bf16 v16bf;
typedef __attribute__((ext_vector_type(8)))  __bf16 v8bf;
typedef __attribute__((ext_vector_type(8)))  float  v8f;

// ---------------------------------------------------------------------------
// fp32 -> bf16 round-to-nearest-even
// ---------------------------------------------------------------------------
__device__ __forceinline__ uint16_t f2bf(float f) {
    uint32_t u = __float_as_uint(f);
    uint32_t r = u + 0x7FFFu + ((u >> 16) & 1u);
    if ((u & 0x7F800000u) == 0x7F800000u) r = u;   // NaN/Inf passthrough
    return (uint16_t)(r >> 16);
}

// ---------------------------------------------------------------------------
// Prep kernels (run once per launch; all grid-stride, deterministic)
// ---------------------------------------------------------------------------
__global__ void cvt_bf16_kernel(const float* __restrict__ src,
                                uint16_t* __restrict__ dst, int n) {
    for (int i = blockIdx.x * blockDim.x + threadIdx.x; i < n;
         i += gridDim.x * blockDim.x)
        dst[i] = f2bf(src[i]);
}

// rows x src_cols fp32  ->  rows x dst_cols bf16, zero padded
__global__ void cvt_pad_kernel(const float* __restrict__ src,
                               uint16_t* __restrict__ dst,
                               int rows, int sc, int dc) {
    int n = rows * dc;
    for (int i = blockIdx.x * blockDim.x + threadIdx.x; i < n;
         i += gridDim.x * blockDim.x) {
        int r = i / dc, c = i - r * dc;
        dst[i] = (c < sc) ? f2bf(src[r * sc + c]) : (uint16_t)0;
    }
}

__global__ void add_bias_kernel(const float* __restrict__ a,
                                const float* __restrict__ b,
                                float* __restrict__ dst, int n) {
    for (int i = blockIdx.x * blockDim.x + threadIdx.x; i < n;
         i += gridDim.x * blockDim.x)
        dst[i] = a[i] + b[i];
}

__global__ void init_state_kernel(const float* __restrict__ hiddens,
                                  const float* __restrict__ cells,
                                  uint16_t* __restrict__ h0b, uint16_t* __restrict__ h1b,
                                  float* __restrict__ h1f,
                                  float* __restrict__ c0, float* __restrict__ c1, int n) {
    for (int i = blockIdx.x * blockDim.x + threadIdx.x; i < n;
         i += gridDim.x * blockDim.x) {
        float hl0 = hiddens[i], hl1 = hiddens[n + i];
        h0b[i] = f2bf(hl0);
        h1b[i] = f2bf(hl1);
        h1f[i] = hl1;
        c0[i]  = cells[i];
        c1[i]  = cells[n + i];
    }
}

// ---------------------------------------------------------------------------
// WMMA fragment load helpers.
//   16-bit A (16x32, MxK): lane m=lane&15; lanes 0-15 hold K {0..7,16..23},
//   lanes 16-31 hold K {8..15,24..31}  -> two contiguous 16B chunks per lane.
//   16-bit B (32x16, KxN): lane n=lane&15; lanes 0-15 hold K {0..7,16..23},
//   lanes 16-31 K {8..15,24..31}       -> two contiguous 16B chunks per lane.
// ---------------------------------------------------------------------------
__device__ __forceinline__ v16bf load_frag2(const bf16_t* p0, const bf16_t* p1) {
    v8bf lo = *(const v8bf*)p0;
    v8bf hi = *(const v8bf*)p1;
    return __builtin_shufflevector(lo, hi, 0, 1, 2, 3, 4, 5, 6, 7,
                                           8, 9, 10, 11, 12, 13, 14, 15);
}

__device__ __forceinline__ void wmma4(const v16bf (&a)[4], v16bf b, v8f* acc) {
#pragma unroll
    for (int m = 0; m < 4; ++m)
        acc[m] = __builtin_amdgcn_wmma_f32_16x16x32_bf16(
            false, a[m], false, b, (short)0, acc[m], false, false);
}

// acc[m] += A[m-tile] (MxK) * W^T (KxN) over k in [kbeg, kend) for one wave's
// 64(M) x 16(N) strip.  A, W row-major with row stride ld (both have K cols).
// Ping-pong software pipeline, unrolled by 2 k-tiles:
//   - every fragment register is defined by a load (no register rotation moves)
//   - 5 moving base pointers + immediate offsets for the two tiles
//   - next tile pair always in flight while current pair's WMMAs issue
__device__ __forceinline__ void gemm_acc(const bf16_t* __restrict__ A,
                                         const bf16_t* __restrict__ W, int ld,
                                         int kbeg, int kend,
                                         int n_base, int lane, v8f* acc) {
    const int ncol = lane & 15;
    const int hl   = lane >> 4;                         // 0 or 1
    const bf16_t* pw  = W + (size_t)(n_base + ncol) * (size_t)ld + kbeg + hl * 8;
    const bf16_t* pa0 = A + (size_t)(ncol)          * (size_t)ld + kbeg + hl * 8;
    const bf16_t* pa1 = A + (size_t)(16 + ncol)     * (size_t)ld + kbeg + hl * 8;
    const bf16_t* pa2 = A + (size_t)(32 + ncol)     * (size_t)ld + kbeg + hl * 8;
    const bf16_t* pa3 = A + (size_t)(48 + ncol)     * (size_t)ld + kbeg + hl * 8;

    // prologue: tile S0 = tile at kbeg (pointer offset 0)
    v16bf b0 = load_frag2(pw, pw + 16);
    v16bf a0[4] = { load_frag2(pa0, pa0 + 16), load_frag2(pa1, pa1 + 16),
                    load_frag2(pa2, pa2 + 16), load_frag2(pa3, pa3 + 16) };
    v16bf b1;
    v16bf a1[4];

    int k = kbeg + 32;
    // invariant at loop head: S0 holds tile (k-32); pointers based at (k-32)
    for (; k + 32 < kend; k += 64) {
        __builtin_prefetch((const void*)(pw + 544), 0, 3);
        // S1 = tile k  (offset +32 elements = +64B)
        b1 = load_frag2(pw + 32, pw + 48);
        a1[0] = load_frag2(pa0 + 32, pa0 + 48);
        a1[1] = load_frag2(pa1 + 32, pa1 + 48);
        a1[2] = load_frag2(pa2 + 32, pa2 + 48);
        a1[3] = load_frag2(pa3 + 32, pa3 + 48);
        wmma4(a0, b0, acc);                      // tile k-32
        pw += 64; pa0 += 64; pa1 += 64; pa2 += 64; pa3 += 64;
        // S0 = tile k+32 (new pointer offset 0)
        b0 = load_frag2(pw, pw + 16);
        a0[0] = load_frag2(pa0, pa0 + 16);
        a0[1] = load_frag2(pa1, pa1 + 16);
        a0[2] = load_frag2(pa2, pa2 + 16);
        a0[3] = load_frag2(pa3, pa3 + 16);
        wmma4(a1, b1, acc);                      // tile k
    }
    if (k < kend) {
        // one trailing tile at k (offset +32 from current base)
        b1 = load_frag2(pw + 32, pw + 48);
        a1[0] = load_frag2(pa0 + 32, pa0 + 48);
        a1[1] = load_frag2(pa1 + 32, pa1 + 48);
        a1[2] = load_frag2(pa2 + 32, pa2 + 48);
        a1[3] = load_frag2(pa3 + 32, pa3 + 48);
        wmma4(a0, b0, acc);
        wmma4(a1, b1, acc);
    } else {
        wmma4(a0, b0, acc);
    }
}

// ---------------------------------------------------------------------------
// Fused LSTM layer step:
//   gates[64, 4H] = A1 @ W1^T + A2 @ W2^T + bias, then pointwise LSTM update.
// grid.x = H/16 = 128 workgroups; 256 threads = 8 waves.
// Wave w: gate = w & 3, k-half = w >> 2.  WG jb owns hidden cols
// [16*jb, 16*jb+16).  Disjoint K halves -> every weight element loaded exactly
// once, 2x outstanding L2 requests per WGP, half the k-chain per wave.
// Partial sums + all 4 gates meet in LDS; c/h updated in-kernel.
// ---------------------------------------------------------------------------
__global__ __launch_bounds__(256)
void lstm_layer_fused(const bf16_t* __restrict__ A1,
                      const bf16_t* __restrict__ W1, int ld1, int K1, int K1h,
                      const bf16_t* __restrict__ A2,
                      const bf16_t* __restrict__ W2, int ld2, int K2, int K2h,
                      const float* __restrict__ bias,
                      float* __restrict__ c,
                      float* __restrict__ h_f32,          // nullptr for layer 0
                      uint16_t* __restrict__ h_bf16) {
    __shared__ float lds_g[2][4][64][17];                // padded: no bank alias

    const int lane = threadIdx.x & 31;
    const int wave = threadIdx.x >> 5;
    const int gate = wave & 3;
    const int kh   = wave >> 2;                          // k-half 0/1
    const int jb   = blockIdx.x * 16;                    // hidden-col base
    const int n_base = gate * HH + jb;                   // col base in 4H space
    const int ncol = lane & 15;
    const int hl   = lane >> 4;

    const float bv = (kh == 0) ? bias[n_base + ncol] : 0.0f;
    v8f acc[4];
#pragma unroll
    for (int m = 0; m < 4; ++m)
#pragma unroll
        for (int v = 0; v < 8; ++v) acc[m][v] = bv;

    const int k1b = kh ? K1h : 0, k1e = kh ? K1 : K1h;
    const int k2b = kh ? K2h : 0, k2e = kh ? K2 : K2h;
    gemm_acc(A1, W1, ld1, k1b, k1e, n_base, lane, acc);
    gemm_acc(A2, W2, ld2, k2b, k2e, n_base, lane, acc);

    // C/D layout: lanes 0-15 -> M = m*16 + v ; lanes 16-31 -> M = m*16 + 8 + v
#pragma unroll
    for (int m = 0; m < 4; ++m) {
        int row0 = m * 16 + hl * 8;
#pragma unroll
        for (int v = 0; v < 8; ++v)
            lds_g[kh][gate][row0 + v][ncol] = acc[m][v];
    }
    __syncthreads();

    // pointwise LSTM: 64 batches x 16 cols = 1024 elems over 256 threads
    for (int e = threadIdx.x; e < BB * 16; e += 256) {
        int b = e >> 4;
        int j = e & 15;
        float gi = lds_g[0][0][b][j] + lds_g[1][0][b][j];
        float gf = lds_g[0][1][b][j] + lds_g[1][1][b][j];
        float gg = lds_g[0][2][b][j] + lds_g[1][2][b][j];
        float go = lds_g[0][3][b][j] + lds_g[1][3][b][j];
        gi = 1.0f / (1.0f + __expf(-gi));
        gf = 1.0f / (1.0f + __expf(-gf));
        go = 1.0f / (1.0f + __expf(-go));
        gg = tanhf(gg);
        int idx = b * HH + (jb + j);
        float c2 = gf * c[idx] + gi * gg;
        c[idx] = c2;
        float h2 = go * tanhf(c2);
        if (h_f32) h_f32[idx] = h2;
        h_bf16[idx] = f2bf(h2);
    }
}

// ---------------------------------------------------------------------------
// out[b,t,:] = clip(h1 @ fc_w^T + fc_b, -1, 1); also feeds x_bf16 for next step
// ---------------------------------------------------------------------------
__global__ void fc_out_kernel(const float* __restrict__ h1,
                              const float* __restrict__ fc_w,
                              const float* __restrict__ fc_b,
                              float* __restrict__ out,
                              uint16_t* __restrict__ x_bf16, int t) {
    int idx = blockIdx.x * blockDim.x + threadIdx.x;
    if (idx >= BB * NOUT) return;
    int b = idx / NOUT, o = idx - b * NOUT;
    const float4* hv = (const float4*)(h1 + (size_t)b * HH);
    const float4* wv = (const float4*)(fc_w + (size_t)o * HH);
    float s = fc_b[o];
    for (int k = 0; k < HH / 4; ++k) {
        float4 a = hv[k], w = wv[k];
        s += a.x * w.x + a.y * w.y + a.z * w.z + a.w * w.w;
    }
    s = fminf(1.0f, fmaxf(-1.0f, s));
    out[((size_t)b * TT + t) * NOUT + o] = s;
    x_bf16[b * KPAD + o] = f2bf(s);
}

// ---------------------------------------------------------------------------
// Host-side orchestration (stream-ordered only; graph-capture safe)
// ---------------------------------------------------------------------------
extern "C" void kernel_launch(void* const* d_in, const int* in_sizes, int n_in,
                              void* d_out, int out_size, void* d_ws, size_t ws_size,
                              hipStream_t stream) {
    (void)in_sizes; (void)n_in; (void)out_size; (void)ws_size;

    const float* inputs  = (const float*)d_in[0];
    const float* hiddens = (const float*)d_in[1];
    const float* cells   = (const float*)d_in[2];
    const float* W_ih0   = (const float*)d_in[3];
    const float* W_hh0   = (const float*)d_in[4];
    const float* b_ih0   = (const float*)d_in[5];
    const float* b_hh0   = (const float*)d_in[6];
    const float* W_ih1   = (const float*)d_in[7];
    const float* W_hh1   = (const float*)d_in[8];
    const float* b_ih1   = (const float*)d_in[9];
    const float* b_hh1   = (const float*)d_in[10];
    const float* fc_w    = (const float*)d_in[11];
    const float* fc_b    = (const float*)d_in[12];
    float* out = (float*)d_out;

    // workspace carve-out (256B aligned regions)
    uintptr_t p = (uintptr_t)d_ws;
    auto alloc = [&](size_t bytes) {
        void* r = (void*)p;
        p += (bytes + 255) & ~(size_t)255;
        return r;
    };
    uint16_t* whh0  = (uint16_t*)alloc((size_t)G4H * HH * 2);
    uint16_t* wih1  = (uint16_t*)alloc((size_t)G4H * HH * 2);
    uint16_t* whh1  = (uint16_t*)alloc((size_t)G4H * HH * 2);
    uint16_t* wih0p = (uint16_t*)alloc((size_t)G4H * KPAD * 2);
    uint16_t* xb    = (uint16_t*)alloc((size_t)BB * KPAD * 2);
    uint16_t* h0b   = (uint16_t*)alloc((size_t)BB * HH * 2);
    uint16_t* h1b   = (uint16_t*)alloc((size_t)BB * HH * 2);
    float*    h1f   = (float*)alloc((size_t)BB * HH * 4);
    float*    c0    = (float*)alloc((size_t)BB * HH * 4);
    float*    c1    = (float*)alloc((size_t)BB * HH * 4);
    float*    bias0 = (float*)alloc((size_t)G4H * 4);
    float*    bias1 = (float*)alloc((size_t)G4H * 4);

    // ---- prep: weight conversion (bf16 weights ~97MB -> L2 resident) ----
    const int nW = G4H * HH;
    cvt_bf16_kernel<<<4096, 256, 0, stream>>>(W_hh0, whh0, nW);
    cvt_bf16_kernel<<<4096, 256, 0, stream>>>(W_ih1, wih1, nW);
    cvt_bf16_kernel<<<4096, 256, 0, stream>>>(W_hh1, whh1, nW);
    cvt_pad_kernel<<<(G4H * KPAD + 255) / 256, 256, 0, stream>>>(W_ih0, wih0p, G4H, NIN, KPAD);
    cvt_pad_kernel<<<(BB * KPAD + 255) / 256, 256, 0, stream>>>(inputs, xb, BB, NIN, KPAD);
    add_bias_kernel<<<(G4H + 255) / 256, 256, 0, stream>>>(b_ih0, b_hh0, bias0, G4H);
    add_bias_kernel<<<(G4H + 255) / 256, 256, 0, stream>>>(b_ih1, b_hh1, bias1, G4H);
    init_state_kernel<<<(BB * HH + 255) / 256, 256, 0, stream>>>(
        hiddens, cells, h0b, h1b, h1f, c0, c1, BB * HH);

    // ---- 25 sequential timesteps, 3 kernels each (stream order = deps) ----
    for (int t = 0; t < TT; ++t) {
        // layer 0: gates = x @ Wih0^T + h0 @ Whh0^T + bias0 ; update c0,h0
        lstm_layer_fused<<<HH / 16, 256, 0, stream>>>(
            (const bf16_t*)xb,  (const bf16_t*)wih0p, KPAD, KPAD, 64,
            (const bf16_t*)h0b, (const bf16_t*)whh0,  HH,   HH,   HH / 2,
            bias0, c0, (float*)nullptr, h0b);

        // layer 1: gates = h0 @ Wih1^T + h1 @ Whh1^T + bias1 ; update c1,h1
        lstm_layer_fused<<<HH / 16, 256, 0, stream>>>(
            (const bf16_t*)h0b, (const bf16_t*)wih1, HH, HH, HH / 2,
            (const bf16_t*)h1b, (const bf16_t*)whh1, HH, HH, HH / 2,
            bias1, c1, h1f, h1b);

        // fc + hardtanh + feedback into next step's x
        fc_out_kernel<<<(BB * NOUT + 127) / 128, 128, 0, stream>>>(
            h1f, fc_w, fc_b, out, xb, t);
    }
}
